// RQQTransformer_88433376625012
// MI455X (gfx1250) — matrix operand order repaired and checked by reading, not compiled, not run on hardware
//
#include <hip/hip_runtime.h>
#include <hip/hip_bf16.h>

// ---------------------------------------------------------------------------
// CDNA5 (gfx1250) WMMA bf16 implementation of the RQQ transformer forward.
// wave32; WMMA 16x16x32 bf16 (f32 accumulate) for all GEMMs and spatial attn.
// ---------------------------------------------------------------------------

typedef __attribute__((ext_vector_type(16))) __bf16 v16bf;
typedef __attribute__((ext_vector_type(8)))  float  v8f;

union AFq { v16bf v; uint4 q[2]; unsigned int u[8]; };

enum : int {
  B_     = 8,
  S_     = 512,
  DEPTH_ = 4,
  DIM_   = 512,
  INNER_ = 512,
  HEADS_ = 8,
  DH_    = 64,
  FFD_   = 2048,
  NTOK_  = 256,
  NSEQ   = 513,             // S+1
  MSP    = B_ * NSEQ,       // 4104 spatial rows
  MDP    = MSP * DEPTH_,    // 16416 depth rows
  VTS    = 544              // padded key-stride for transposed V
};

// ---------------------------------------------------------------------------
// WMMA GEMM: C[M,N] = A[M,K](bf16, row stride lda) x Bt[N,K](bf16 = B^T).
// One wave computes a 32x64 C tile (2 row-groups x 4 N-tiles; B fragments
// reused across both row groups). K pipelined in 32-deep steps with two
// fragment sets (requires K % 64 == 0). Out-of-range A rows are clamped
// (their C rows are never stored).
// Epilogue: +bias, optional exact GELU, +residual, f32 and/or bf16 stores.
// ---------------------------------------------------------------------------
__global__ __launch_bounds__(32)
void gemm_bf16_kernel(const __bf16* __restrict__ A, long lda,
                      const __bf16* __restrict__ Bt,
                      const float* __restrict__ bias,
                      const float* __restrict__ resid, long ldr,
                      float* __restrict__ Cf, long ldcf,
                      __bf16* __restrict__ Cb, long ldcb,
                      int M, int N, int K, int gelu)
{
  const int tm   = blockIdx.x;
  const int tn0  = blockIdx.y * 4;
  const int lane = threadIdx.x;
  const int li   = lane & 15;
  const int hi   = (lane >> 4) & 1;

  int arow0 = tm * 32 + li;
  int arow1 = tm * 32 + 16 + li;
  if (arow0 >= M) arow0 = M - 1;                     // clamp: no divergent loads
  if (arow1 >= M) arow1 = M - 1;
  const __bf16* Arow0 = A + (long)arow0 * lda + hi * 8;
  const __bf16* Arow1 = A + (long)arow1 * lda + hi * 8;
  const __bf16* Brow  = Bt + (long)(tn0 * 16 + li) * K + hi * 8;

  v8f acc[2][4];
#pragma unroll
  for (int g = 0; g < 2; ++g)
#pragma unroll
    for (int t = 0; t < 4; ++t)
      acc[g][t] = (v8f){0.f,0.f,0.f,0.f,0.f,0.f,0.f,0.f};

  AFq a0[2], a1[2], b0[4], b1[4];

  // A fragment: two contiguous 16B spans; B fragment: two contiguous 16B spans.
#define LOAD_A(dst, k0)                                              \
  do {                                                               \
    (dst)[0].q[0] = *(const uint4*)(Arow0 + (k0));                   \
    (dst)[0].q[1] = *(const uint4*)(Arow0 + (k0) + 16);              \
    (dst)[1].q[0] = *(const uint4*)(Arow1 + (k0));                   \
    (dst)[1].q[1] = *(const uint4*)(Arow1 + (k0) + 16);              \
  } while (0)
#define LOAD_B(dst, k0)                                              \
  do {                                                               \
    _Pragma("unroll")                                                \
    for (int t = 0; t < 4; ++t) {                                    \
      const __bf16* bp = Brow + (long)t * 16 * K + (k0);             \
      (dst)[t].q[0] = *(const uint4*)(bp);                           \
      (dst)[t].q[1] = *(const uint4*)(bp + 16);                      \
    }                                                                \
  } while (0)

  LOAD_A(a0, 0);
  LOAD_B(b0, 0);
  for (int k0 = 0; k0 < K; k0 += 64) {
    LOAD_A(a1, k0 + 32);
    LOAD_B(b1, k0 + 32);
#pragma unroll
    for (int t = 0; t < 4; ++t) {
      acc[0][t] = __builtin_amdgcn_wmma_f32_16x16x32_bf16(false, a0[0].v, false, b0[t].v,
                                                          (short)0, acc[0][t], false, false);
      acc[1][t] = __builtin_amdgcn_wmma_f32_16x16x32_bf16(false, a0[1].v, false, b0[t].v,
                                                          (short)0, acc[1][t], false, false);
    }
    if (k0 + 64 < K) {
      LOAD_A(a0, k0 + 64);
      LOAD_B(b0, k0 + 64);
    }
#pragma unroll
    for (int t = 0; t < 4; ++t) {
      acc[0][t] = __builtin_amdgcn_wmma_f32_16x16x32_bf16(false, a1[0].v, false, b1[t].v,
                                                          (short)0, acc[0][t], false, false);
      acc[1][t] = __builtin_amdgcn_wmma_f32_16x16x32_bf16(false, a1[1].v, false, b1[t].v,
                                                          (short)0, acc[1][t], false, false);
    }
  }
#undef LOAD_A
#undef LOAD_B

#pragma unroll
  for (int t = 0; t < 4; ++t) {
    const int ncol = (tn0 + t) * 16 + li;
    const float bv = bias ? bias[ncol] : 0.f;
#pragma unroll
    for (int g = 0; g < 2; ++g) {
#pragma unroll
      for (int r = 0; r < 8; ++r) {
        const int row = tm * 32 + g * 16 + r + hi * 8;
        if (row >= M) continue;
        float val = acc[g][t][r] + bv;
        if (gelu) val = 0.5f * val * (1.f + erff(val * 0.70710678118654752f));
        if (resid) val += resid[(long)row * ldr + ncol];
        if (Cf) Cf[(long)row * ldcf + ncol] = val;
        if (Cb) Cb[(long)row * ldcb + ncol] = (__bf16)val;
      }
    }
  }
}

// ---------------------------------------------------------------------------
// LayerNorm over DIM=512, one wave per row (16 contiguous floats per lane).
// ---------------------------------------------------------------------------
__global__ __launch_bounds__(32)
void layernorm_kernel(const float* __restrict__ X,
                      const float* __restrict__ g, const float* __restrict__ bta,
                      float* __restrict__ Of, __bf16* __restrict__ Ob, int M)
{
  const int row = blockIdx.x;
  if (row >= M) return;
  const int lane = threadIdx.x;
  const float* x = X + (long)row * DIM_;

  float4 vv[4];
  float s = 0.f, s2 = 0.f;
#pragma unroll
  for (int j = 0; j < 4; ++j) {
    vv[j] = ((const float4*)x)[lane * 4 + j];
    s  += vv[j].x + vv[j].y + vv[j].z + vv[j].w;
    s2 += vv[j].x*vv[j].x + vv[j].y*vv[j].y + vv[j].z*vv[j].z + vv[j].w*vv[j].w;
  }
#pragma unroll
  for (int off = 16; off >= 1; off >>= 1) {
    s  += __shfl_xor(s,  off, 32);
    s2 += __shfl_xor(s2, off, 32);
  }
  const float mean = s * (1.f / DIM_);
  const float var  = s2 * (1.f / DIM_) - mean * mean;
  const float inv  = rsqrtf(var + 1e-5f);

#pragma unroll
  for (int j = 0; j < 4; ++j) {
    const float vj[4] = {vv[j].x, vv[j].y, vv[j].z, vv[j].w};
#pragma unroll
    for (int e = 0; e < 4; ++e) {
      const int c = lane * 16 + j * 4 + e;
      const float y = (vj[e] - mean) * inv * g[c] + bta[c];
      if (Of) Of[(long)row * DIM_ + c] = y;
      if (Ob) Ob[(long)row * DIM_ + c] = (__bf16)y;
    }
  }
}

// ---------------------------------------------------------------------------
// Weight transpose f32[K][N] -> bf16[N][K], LDS-tiled (both sides coalesced).
// K, N multiples of 32.
// ---------------------------------------------------------------------------
__global__ __launch_bounds__(256)
void wtrans_kernel(const float* __restrict__ W, __bf16* __restrict__ Wt,
                   int K, int N)
{
  __shared__ float tile[32][33];
  const int n0 = blockIdx.x * 32;
  const int k0 = blockIdx.y * 32;
  const int tx = threadIdx.x;      // 0..31
  const int ty = threadIdx.y;      // 0..7
#pragma unroll
  for (int i = ty; i < 32; i += 8)
    tile[i][tx] = W[(long)(k0 + i) * N + (n0 + tx)];
  __syncthreads();
#pragma unroll
  for (int i = ty; i < 32; i += 8)
    Wt[(long)(n0 + i) * K + (k0 + tx)] = (__bf16)tile[tx][i];
}

// ---------------------------------------------------------------------------
// Embedding: spatial[b][s][c] = sum_d(tok_emb[ids]+dp_pos) + sp_pos[s]
// ---------------------------------------------------------------------------
__global__ void embed_kernel(const int* __restrict__ ids,
                             const float* __restrict__ tok_emb,
                             const float* __restrict__ sp_pos,
                             const float* __restrict__ dp_pos,
                             float* __restrict__ out)
{
  const long idx = (long)blockIdx.x * blockDim.x + threadIdx.x;
  if (idx >= (long)B_ * S_ * DIM_) return;
  const int c = (int)(idx & (DIM_ - 1));
  const int s = (int)((idx >> 9) & (S_ - 1));
  const int b = (int)(idx >> 18);
  const int* id4 = ids + ((long)b * S_ + s) * DEPTH_;
  float acc = sp_pos[(long)s * DIM_ + c];
#pragma unroll
  for (int d = 0; d < DEPTH_; ++d)
    acc += tok_emb[(long)id4[d] * DIM_ + c] + dp_pos[(long)d * DIM_ + c];
  out[idx] = acc;
}

// ---------------------------------------------------------------------------
// Mask prep: sp_mask (all-depth AND), mask_ext (append 1), lpi (#left pads).
// ---------------------------------------------------------------------------
__global__ void mask_kernel(const unsigned char* __restrict__ am,
                            unsigned char* __restrict__ spmask,
                            unsigned char* __restrict__ maskext)
{
  const int b = blockIdx.x;
  const int s = threadIdx.x;
  const unsigned char* a = am + ((long)b * S_ + s) * DEPTH_;
  const unsigned char m = (a[0] && a[1] && a[2] && a[3]) ? 1 : 0;
  spmask[(long)b * S_ + s]  = m;
  maskext[(long)b * NSEQ + s] = m;
  if (s == 0) maskext[(long)b * NSEQ + S_] = 1;
}

__global__ void lpi_kernel(const unsigned char* __restrict__ spmask,
                           int* __restrict__ lpi)
{
  const int b = threadIdx.x;
  if (b >= B_) return;
  int c = 0;
  for (int s = 0; s < S_; ++s) c += spmask[(long)b * S_ + s] ? 0 : 1;
  lpi[b] = c;
}

// ---------------------------------------------------------------------------
// Start-token insertion (left-padding aware, mirrors reference priority).
// ---------------------------------------------------------------------------
__global__ void insert_kernel(const float* __restrict__ xin,   // [B][S][DIM]
                              const float* __restrict__ start, // [DIM]
                              const unsigned char* __restrict__ spmask,
                              const int* __restrict__ lpi,
                              float* __restrict__ xout)        // [B][NSEQ][DIM]
{
  const long idx = (long)blockIdx.x * blockDim.x + threadIdx.x;
  if (idx >= (long)B_ * NSEQ * DIM_) return;
  const int c = (int)(idx & (DIM_ - 1));
  const int p = (int)((idx >> 9) % NSEQ);
  const int b = (int)(idx / ((long)NSEQ * DIM_));
  float v;
  if (p >= 1 && spmask[(long)b * S_ + p - 1])
    v = xin[((long)b * S_ + p - 1) * DIM_ + c];
  else if (p < S_ && !spmask[(long)b * S_ + p])
    v = xin[((long)b * S_ + p) * DIM_ + c];
  else if (p == lpi[b])
    v = start[c];
  else
    v = 0.f;
  xout[idx] = v;
}

// ---------------------------------------------------------------------------
// QKV reorg for spatial attention:
//   Q[b][h][row][64] (scaled 1/8), K[b][h][row][64], Vt[b][h][64][VTS] (pad 0)
// ---------------------------------------------------------------------------
__global__ void qkv_reorg_kernel(const __bf16* __restrict__ qkv, // [MSP][1536]
                                 __bf16* __restrict__ Q,
                                 __bf16* __restrict__ Kt,
                                 __bf16* __restrict__ Vt)
{
  const long idx = (long)blockIdx.x * blockDim.x + threadIdx.x;
  if (idx >= (long)B_ * HEADS_ * VTS * DH_) return;
  const int dim = (int)(idx & (DH_ - 1));
  const int row = (int)((idx >> 6) % VTS);
  const int h   = (int)((idx / ((long)VTS * DH_)) % HEADS_);
  const int b   = (int)(idx / ((long)HEADS_ * VTS * DH_));
  const long bh = (long)b * HEADS_ + h;
  if (row < NSEQ) {
    const __bf16* src = qkv + ((long)b * NSEQ + row) * (3 * INNER_) + h * DH_ + dim;
    const __bf16 qv = src[0];
    const __bf16 kv = src[INNER_];
    const __bf16 vv = src[2 * INNER_];
    Q [(bh * NSEQ + row) * DH_ + dim] = (__bf16)((float)qv * 0.125f);
    Kt[(bh * NSEQ + row) * DH_ + dim] = kv;
    Vt[(bh * DH_ + dim) * VTS + row]  = vv;
  } else {
    Vt[(bh * DH_ + dim) * VTS + row]  = (__bf16)0.f;
  }
}

// ---------------------------------------------------------------------------
// Spatial flash-attention with WMMA. One wave per (qtile, head, batch).
// All fragment loads are contiguous 16B spans; row/key indices are clamped
// (garbage only lands in rows/keys that are masked or never stored).
// ---------------------------------------------------------------------------
__global__ __launch_bounds__(32)
void sp_attn_kernel(const __bf16* __restrict__ Q,
                    const __bf16* __restrict__ Kt,
                    const __bf16* __restrict__ Vt,
                    const unsigned char* __restrict__ maskext,
                    __bf16* __restrict__ O)     // [B*NSEQ][INNER]
{
  __shared__ float pbuf[16][32];

  const int qt = blockIdx.x, h = blockIdx.y, b = blockIdx.z;
  const int lane = threadIdx.x;
  const int li   = lane & 15;
  const int hi   = (lane >> 4) & 1;

  const long bh = (long)b * HEADS_ + h;
  const __bf16* Qbh = Q  + bh * NSEQ * DH_;
  const __bf16* Kbh = Kt + bh * NSEQ * DH_;
  const __bf16* Vbh = Vt + bh * DH_ * VTS;
  const unsigned char* me = maskext + (long)b * NSEQ;

  // Q A-fragments for the two K-dim steps (0..31, 32..63)
  AFq aq[2];
  {
    int qrow_l = qt * 16 + li;
    if (qrow_l >= NSEQ) qrow_l = NSEQ - 1;     // clamp
    const __bf16* qp = Qbh + (long)qrow_l * DH_ + hi * 8;
#pragma unroll
    for (int s = 0; s < 2; ++s) {
      aq[s].q[0] = *(const uint4*)(qp + s * 32);
      aq[s].q[1] = *(const uint4*)(qp + s * 32 + 16);
    }
  }

  v8f o0 = {0,0,0,0,0,0,0,0}, o1 = o0, o2 = o0, o3 = o0;
  float mi[8], lsum[8];
#pragma unroll
  for (int r = 0; r < 8; ++r) { mi[r] = -1e30f; lsum[r] = 0.f; }

  const int qmax = (qt * 16 + 15 < NSEQ - 1) ? (qt * 16 + 15) : (NSEQ - 1);

  for (int j0 = 0; j0 <= qmax; j0 += 32) {
    const int keyl = j0 + li;
    const int keyh = j0 + 16 + li;
    const int keylc = keyl < NSEQ ? keyl : NSEQ - 1;
    const int keyhc = keyh < NSEQ ? keyh : NSEQ - 1;
    const __bf16* kpl = Kbh + (long)keylc * DH_ + hi * 8;
    const __bf16* kph = Kbh + (long)keyhc * DH_ + hi * 8;

    v8f slo = {0,0,0,0,0,0,0,0}, shi = slo;
#pragma unroll
    for (int s = 0; s < 2; ++s) {
      AFq bl, bh2;
      bl.q[0]  = *(const uint4*)(kpl + s * 32);
      bl.q[1]  = *(const uint4*)(kpl + s * 32 + 16);
      bh2.q[0] = *(const uint4*)(kph + s * 32);
      bh2.q[1] = *(const uint4*)(kph + s * 32 + 16);
      slo = __builtin_amdgcn_wmma_f32_16x16x32_bf16(false, aq[s].v, false, bl.v,
                                                    (short)0, slo, false, false);
      shi = __builtin_amdgcn_wmma_f32_16x16x32_bf16(false, aq[s].v, false, bh2.v,
                                                    (short)0, shi, false, false);
    }

    const bool okl = (keyl < NSEQ) && me[keylc];
    const bool okh = (keyh < NSEQ) && me[keyhc];

    float corr[8], plo[8], phi[8];
#pragma unroll
    for (int r = 0; r < 8; ++r) {
      const int qr = qt * 16 + r + hi * 8;
      float sl = (okl && keyl <= qr && qr < NSEQ) ? slo[r] : -1e30f;
      float sh = (okh && keyh <= qr && qr < NSEQ) ? shi[r] : -1e30f;
      float t = fmaxf(sl, sh);
#pragma unroll
      for (int off = 8; off >= 1; off >>= 1) t = fmaxf(t, __shfl_xor(t, off, 32));
      const float nm = fmaxf(mi[r], t);
      const float c  = __expf(mi[r] - nm);
      const float pl = __expf(sl - nm);
      const float ph = __expf(sh - nm);
      float rs = pl + ph;
#pragma unroll
      for (int off = 8; off >= 1; off >>= 1) rs += __shfl_xor(rs, off, 32);
      lsum[r] = lsum[r] * c + rs;
      mi[r] = nm; corr[r] = c; plo[r] = pl; phi[r] = ph;
    }

#pragma unroll
    for (int r = 0; r < 8; ++r) {
      o0[r] *= corr[r]; o1[r] *= corr[r]; o2[r] *= corr[r]; o3[r] *= corr[r];
    }

    // C-layout P -> LDS -> A-layout bf16
#pragma unroll
    for (int r = 0; r < 8; ++r) {
      pbuf[r + hi * 8][li]      = plo[r];
      pbuf[r + hi * 8][16 + li] = phi[r];
    }
    __syncthreads();
    AFq ap;
#pragma unroll
    for (int v = 0; v < 8; ++v) {
      const int k = ((v < 4) ? (2 * v) : (16 + 2 * (v - 4))) + hi * 8;
      const float2 p2 = *(const float2*)&pbuf[li][k];
      ap.v[2 * v]     = (__bf16)p2.x;
      ap.v[2 * v + 1] = (__bf16)p2.y;
    }
    __syncthreads();

    // P (16x32) x V (32x64) as 4 N-tiles
#pragma unroll
    for (int t = 0; t < 4; ++t) {
      AFq bv;
      const __bf16* vp = Vbh + (long)(t * 16 + li) * VTS + j0 + hi * 8;
      bv.q[0] = *(const uint4*)(vp);
      bv.q[1] = *(const uint4*)(vp + 16);
      v8f* oacc = (t == 0) ? &o0 : (t == 1) ? &o1 : (t == 2) ? &o2 : &o3;
      *oacc = __builtin_amdgcn_wmma_f32_16x16x32_bf16(false, ap.v, false, bv.v,
                                                      (short)0, *oacc, false, false);
    }
  }

#pragma unroll
  for (int r = 0; r < 8; ++r) {
    const int qr = qt * 16 + r + hi * 8;
    if (qr >= NSEQ) continue;
    const float inv = 1.f / lsum[r];
    const long base = ((long)b * NSEQ + qr) * INNER_ + h * DH_;
    O[base +  0 + li] = (__bf16)(o0[r] * inv);
    O[base + 16 + li] = (__bf16)(o1[r] * inv);
    O[base + 32 + li] = (__bf16)(o2[r] * inv);
    O[base + 48 + li] = (__bf16)(o3[r] * inv);
  }
}

// ---------------------------------------------------------------------------
// dt[b,pos,d,:] = lnf_spatial[b,pos,:] + depth_queries[d,:]
// ---------------------------------------------------------------------------
__global__ void build_dt_kernel(const float* __restrict__ xl,  // [MSP][DIM]
                                const float* __restrict__ dq,  // [DEPTH][DIM]
                                float* __restrict__ dt)        // [MDP][DIM]
{
  const long idx = (long)blockIdx.x * blockDim.x + threadIdx.x;
  if (idx >= (long)MDP * DIM_) return;
  const int c = (int)(idx & (DIM_ - 1));
  const int d = (int)((idx >> 9) & (DEPTH_ - 1));
  const long pos = idx / ((long)DEPTH_ * DIM_);
  dt[idx] = xl[pos * DIM_ + c] + dq[(long)d * DIM_ + c];
}

// ---------------------------------------------------------------------------
// Depth attention (seq len 4, full attention). Tiny FLOPs -> plain VALU.
// One thread per (seq, head, query).
// ---------------------------------------------------------------------------
__global__ void dp_attn_kernel(const __bf16* __restrict__ qkv, // [MDP][1536]
                               __bf16* __restrict__ O)         // [MDP][INNER]
{
  const long t = (long)blockIdx.x * blockDim.x + threadIdx.x;
  if (t >= (long)MSP * HEADS_ * DEPTH_) return;
  const int tq  = (int)(t & 3);
  const int h   = (int)((t >> 2) & 7);
  const long seq = t >> 5;
  const __bf16* base = qkv + seq * DEPTH_ * (3 * INNER_);

  float sim[DEPTH_];
#pragma unroll
  for (int tk = 0; tk < DEPTH_; ++tk) {
    float dsum = 0.f;
    for (int d = 0; d < DH_; ++d)
      dsum += (float)base[(long)tq * 1536 + h * DH_ + d] *
              (float)base[(long)tk * 1536 + INNER_ + h * DH_ + d];
    sim[tk] = dsum * 0.125f;
  }
  float mx = sim[0];
#pragma unroll
  for (int tk = 1; tk < DEPTH_; ++tk) mx = fmaxf(mx, sim[tk]);
  float ssum = 0.f, a[DEPTH_];
#pragma unroll
  for (int tk = 0; tk < DEPTH_; ++tk) { a[tk] = __expf(sim[tk] - mx); ssum += a[tk]; }
  const float inv = 1.f / ssum;
#pragma unroll
  for (int tk = 0; tk < DEPTH_; ++tk) a[tk] *= inv;

  for (int d = 0; d < DH_; ++d) {
    float o = 0.f;
#pragma unroll
    for (int tk = 0; tk < DEPTH_; ++tk)
      o += a[tk] * (float)base[(long)tk * 1536 + 2 * INNER_ + h * DH_ + d];
    O[(seq * DEPTH_ + tq) * INNER_ + h * DH_ + d] = (__bf16)o;
  }
}

// ---------------------------------------------------------------------------
// Host launch
// ---------------------------------------------------------------------------
extern "C" void kernel_launch(void* const* d_in, const int* in_sizes, int n_in,
                              void* d_out, int out_size, void* d_ws, size_t ws_size,
                              hipStream_t stream) {
  (void)in_sizes; (void)n_in; (void)out_size; (void)ws_size;

  const int*            ids     = (const int*)d_in[0];
  const unsigned char*  amask   = (const unsigned char*)d_in[1];
  const float*          tok_emb = (const float*)d_in[2];
  const float*          sp_pos  = (const float*)d_in[3];
  const float*          dp_pos  = (const float*)d_in[4];
  const float*          start_t = (const float*)d_in[5];
  const float*          depth_q = (const float*)d_in[6];
  const float* sp_ln1_g = (const float*)d_in[7];
  const float* sp_ln1_b = (const float*)d_in[8];
  const float* sp_wqkv  = (const float*)d_in[9];
  const float* sp_wout  = (const float*)d_in[10];
  const float* sp_ln2_g = (const float*)d_in[11];
  const float* sp_ln2_b = (const float*)d_in[12];
  const float* sp_w1    = (const float*)d_in[13];
  const float* sp_b1    = (const float*)d_in[14];
  const float* sp_w2    = (const float*)d_in[15];
  const float* sp_b2    = (const float*)d_in[16];
  const float* sp_lnf_g = (const float*)d_in[17];
  const float* sp_lnf_b = (const float*)d_in[18];
  const float* dp_ln1_g = (const float*)d_in[19];
  const float* dp_ln1_b = (const float*)d_in[20];
  const float* dp_wqkv  = (const float*)d_in[21];
  const float* dp_wout  = (const float*)d_in[22];
  const float* dp_ln2_g = (const float*)d_in[23];
  const float* dp_ln2_b = (const float*)d_in[24];
  const float* dp_w1    = (const float*)d_in[25];
  const float* dp_b1    = (const float*)d_in[26];
  const float* dp_w2    = (const float*)d_in[27];
  const float* dp_b2    = (const float*)d_in[28];
  const float* dp_lnf_g = (const float*)d_in[29];
  const float* dp_lnf_b = (const float*)d_in[30];
  const float* logits_w = (const float*)d_in[31];
  float* out = (float*)d_out;

  // -------- workspace carving (deterministic) --------
  char* ws = (char*)d_ws;
  size_t off = 0;
  auto take = [&](size_t bytes) -> char* {
    char* p = ws + off;
    off += (bytes + 255) & ~(size_t)255;
    return p;
  };
  __bf16* w_sp_qkv = (__bf16*)take((size_t)4 * 3 * INNER_ * DIM_ * 2);
  __bf16* w_sp_out = (__bf16*)take((size_t)4 * DIM_ * INNER_ * 2);
  __bf16* w_sp_w1  = (__bf16*)take((size_t)4 * FFD_ * DIM_ * 2);
  __bf16* w_sp_w2  = (__bf16*)take((size_t)4 * DIM_ * FFD_ * 2);
  __bf16* w_dp_qkv = (__bf16*)take((size_t)2 * 3 * INNER_ * DIM_ * 2);
  __bf16* w_dp_out = (__bf16*)take((size_t)2 * DIM_ * INNER_ * 2);
  __bf16* w_dp_w1  = (__bf16*)take((size_t)2 * FFD_ * DIM_ * 2);
  __bf16* w_dp_w2  = (__bf16*)take((size_t)2 * DIM_ * FFD_ * 2);
  __bf16* w_log    = (__bf16*)take((size_t)4 * NTOK_ * DIM_ * 2);
  float*  xa       = (float*) take((size_t)MDP * DIM_ * 4);
  float*  xb       = (float*) take((size_t)MDP * DIM_ * 4);
  __bf16* lnb      = (__bf16*)take((size_t)MDP * DIM_ * 2);
  __bf16* qkvb     = (__bf16*)take((size_t)MDP * 3 * INNER_ * 2);
  char*   midraw   =          take((size_t)MDP * FFD_ * 2);
  __bf16* midb     = (__bf16*)midraw;
  // attention scratch carved inside midb (disjoint in time from FFN use)
  __bf16* Qb  = (__bf16*)(midraw);
  __bf16* Kb  = Qb  + (size_t)B_ * HEADS_ * NSEQ * DH_;
  __bf16* Vtb = Kb  + (size_t)B_ * HEADS_ * NSEQ * DH_;
  __bf16* Ab  = Vtb + (size_t)B_ * HEADS_ * DH_ * VTS;     // spatial attn output
  unsigned char* spmask  = (unsigned char*)take((size_t)B_ * S_);
  unsigned char* maskext = (unsigned char*)take((size_t)B_ * NSEQ);
  int* lpi = (int*)take((size_t)B_ * 4);

  auto wt = [&](const float* W, __bf16* Wt, int K, int N) {
    wtrans_kernel<<<dim3(N / 32, K / 32), dim3(32, 8), 0, stream>>>(W, Wt, K, N);
  };
  auto gemm = [&](const __bf16* A, long lda, const __bf16* Bt, const float* bias,
                  const float* resid, long ldr, float* Cf, long ldcf,
                  __bf16* Cb, long ldcb, int M, int N, int K, int gelu) {
    gemm_bf16_kernel<<<dim3((M + 31) / 32, N / 64), 32, 0, stream>>>(
        A, lda, Bt, bias, resid, ldr, Cf, ldcf, Cb, ldcb, M, N, K, gelu);
  };
  auto ln = [&](const float* X, const float* g, const float* b,
                float* Of, __bf16* Ob, int M) {
    layernorm_kernel<<<dim3(M), 32, 0, stream>>>(X, g, b, Of, Ob, M);
  };

  // -------- weights -> bf16 transposed --------
  for (int l = 0; l < 4; ++l) {
    wt(sp_wqkv + (long)l * DIM_ * 3 * INNER_, w_sp_qkv + (long)l * 3 * INNER_ * DIM_, DIM_, 3 * INNER_);
    wt(sp_wout + (long)l * INNER_ * DIM_,     w_sp_out + (long)l * DIM_ * INNER_,     INNER_, DIM_);
    wt(sp_w1   + (long)l * DIM_ * FFD_,       w_sp_w1  + (long)l * FFD_ * DIM_,       DIM_, FFD_);
    wt(sp_w2   + (long)l * FFD_ * DIM_,       w_sp_w2  + (long)l * DIM_ * FFD_,       FFD_, DIM_);
  }
  for (int l = 0; l < 2; ++l) {
    wt(dp_wqkv + (long)l * DIM_ * 3 * INNER_, w_dp_qkv + (long)l * 3 * INNER_ * DIM_, DIM_, 3 * INNER_);
    wt(dp_wout + (long)l * INNER_ * DIM_,     w_dp_out + (long)l * DIM_ * INNER_,     INNER_, DIM_);
    wt(dp_w1   + (long)l * DIM_ * FFD_,       w_dp_w1  + (long)l * FFD_ * DIM_,       DIM_, FFD_);
    wt(dp_w2   + (long)l * FFD_ * DIM_,       w_dp_w2  + (long)l * DIM_ * FFD_,       FFD_, DIM_);
  }
  for (int d = 0; d < 4; ++d)
    wt(logits_w + (long)d * DIM_ * NTOK_, w_log + (long)d * NTOK_ * DIM_, DIM_, NTOK_);

  // -------- embedding, mask, start-token insertion --------
  {
    const long tot = (long)B_ * S_ * DIM_;
    embed_kernel<<<dim3((unsigned)((tot + 255) / 256)), 256, 0, stream>>>(
        ids, tok_emb, sp_pos, dp_pos, xb);
    mask_kernel<<<dim3(B_), S_, 0, stream>>>(amask, spmask, maskext);
    lpi_kernel<<<dim3(1), 32, 0, stream>>>(spmask, lpi);
    const long tot2 = (long)B_ * NSEQ * DIM_;
    insert_kernel<<<dim3((unsigned)((tot2 + 255) / 256)), 256, 0, stream>>>(
        xb, start_t, spmask, lpi, xa);
  }

  // -------- spatial transformer (4 layers, causal) --------
  for (int l = 0; l < 4; ++l) {
    ln(xa, sp_ln1_g + (long)l * DIM_, sp_ln1_b + (long)l * DIM_, nullptr, lnb, MSP);
    gemm(lnb, DIM_, w_sp_qkv + (long)l * 3 * INNER_ * DIM_, nullptr,
         nullptr, 0, nullptr, 0, qkvb, 3 * INNER_, MSP, 3 * INNER_, DIM_, 0);
    {
      const long tot = (long)B_ * HEADS_ * VTS * DH_;
      qkv_reorg_kernel<<<dim3((unsigned)((tot + 255) / 256)), 256, 0, stream>>>(
          qkvb, Qb, Kb, Vtb);
    }
    sp_attn_kernel<<<dim3((NSEQ + 15) / 16, HEADS_, B_), 32, 0, stream>>>(
        Qb, Kb, Vtb, maskext, Ab);
    gemm(Ab, INNER_, w_sp_out + (long)l * DIM_ * INNER_, nullptr,
         xa, DIM_, xb, DIM_, nullptr, 0, MSP, DIM_, INNER_, 0);
    ln(xb, sp_ln2_g + (long)l * DIM_, sp_ln2_b + (long)l * DIM_, nullptr, lnb, MSP);
    gemm(lnb, DIM_, w_sp_w1 + (long)l * FFD_ * DIM_, sp_b1 + (long)l * FFD_,
         nullptr, 0, nullptr, 0, midb, FFD_, MSP, FFD_, DIM_, 1);
    gemm(midb, FFD_, w_sp_w2 + (long)l * DIM_ * FFD_, sp_b2 + (long)l * DIM_,
         xb, DIM_, xa, DIM_, nullptr, 0, MSP, DIM_, FFD_, 0);
  }
  ln(xa, sp_lnf_g, sp_lnf_b, xb, nullptr, MSP);

  // -------- depth tokens --------
  {
    const long tot = (long)MDP * DIM_;
    build_dt_kernel<<<dim3((unsigned)((tot + 255) / 256)), 256, 0, stream>>>(
        xb, depth_q, xa);
  }

  // -------- depth transformer (2 layers, full attention over 4) --------
  for (int l = 0; l < 2; ++l) {
    ln(xa, dp_ln1_g + (long)l * DIM_, dp_ln1_b + (long)l * DIM_, nullptr, lnb, MDP);
    gemm(lnb, DIM_, w_dp_qkv + (long)l * 3 * INNER_ * DIM_, nullptr,
         nullptr, 0, nullptr, 0, qkvb, 3 * INNER_, MDP, 3 * INNER_, DIM_, 0);
    {
      const long tot = (long)MSP * HEADS_ * DEPTH_;
      dp_attn_kernel<<<dim3((unsigned)((tot + 255) / 256)), 256, 0, stream>>>(
          qkvb, lnb);   // lnb reused as depth attn output (bf16)
    }
    gemm(lnb, INNER_, w_dp_out + (long)l * DIM_ * INNER_, nullptr,
         xa, DIM_, xb, DIM_, nullptr, 0, MDP, DIM_, INNER_, 0);
    ln(xb, dp_ln2_g + (long)l * DIM_, dp_ln2_b + (long)l * DIM_, nullptr, lnb, MDP);
    gemm(lnb, DIM_, w_dp_w1 + (long)l * FFD_ * DIM_, dp_b1 + (long)l * FFD_,
         nullptr, 0, nullptr, 0, midb, FFD_, MDP, FFD_, DIM_, 1);
    gemm(midb, FFD_, w_dp_w2 + (long)l * DIM_ * FFD_, dp_b2 + (long)l * DIM_,
         xb, DIM_, xa, DIM_, nullptr, 0, MDP, DIM_, FFD_, 0);
  }
  ln(xa, dp_lnf_g, dp_lnf_b, nullptr, lnb, MDP);

  // -------- per-depth logits heads (strided GEMMs) --------
  for (int d = 0; d < 4; ++d) {
    gemm(lnb + (long)d * DIM_, (long)DEPTH_ * DIM_,
         w_log + (long)d * NTOK_ * DIM_, nullptr,
         nullptr, 0,
         out + (long)d * NTOK_, (long)DEPTH_ * NTOK_,
         nullptr, 0, MSP, NTOK_, DIM_, 0);
  }
}
